// ImprovedBoundingBoxProcessor2_45801531245115
// MI455X (gfx1250) — compile-verified, matrix operand order checked by reading, not run on hardware
//
#include <hip/hip_runtime.h>
#include <hip/hip_bf16.h>

#define N_ANCHORS 6144
#define NCLS 80
#define PAD 8192
#define T 1024

typedef float v2f  __attribute__((ext_vector_type(2)));
typedef float v8f  __attribute__((ext_vector_type(8)));
typedef _Float16 v16h __attribute__((ext_vector_type(16)));

// ---------------- Kernel 1: per-anchor max score + class-0 score ----------------
__global__ __launch_bounds__(256) void score_kernel(const float* __restrict__ conf,
                                                    float* __restrict__ smax,
                                                    float* __restrict__ s0) {
  int a = blockIdx.x * blockDim.x + threadIdx.x;
  if (a >= N_ANCHORS) return;
  const float4* r = (const float4*)(conf + (size_t)a * NCLS);
  // stream the next row through the gfx1250 prefetch path
  __builtin_prefetch(conf + (size_t)(a + 256) * NCLS, 0, 1);
  float4 v = r[0];
  float c0 = v.x;
  float m = fmaxf(fmaxf(v.x, v.y), fmaxf(v.z, v.w));
#pragma unroll
  for (int i = 1; i < NCLS / 4; ++i) {
    float4 w = r[i];
    m = fmaxf(m, fmaxf(fmaxf(w.x, w.y), fmaxf(w.z, w.w)));
  }
  smax[a] = m;
  s0[a] = c0;
}

// Inclusive Hillis-Steele scan over T entries; returns exclusive prefix, writes total.
__device__ __forceinline__ int block_scan(int* scan, int tid, int cnt, int* total) {
  scan[tid] = cnt;
  __syncthreads();
  for (int off = 1; off < T; off <<= 1) {
    int v = scan[tid];
    int add = (tid >= off) ? scan[tid - off] : 0;
    __syncthreads();
    scan[tid] = v + add;
    __syncthreads();
  }
  *total = scan[T - 1];
  return scan[tid] - cnt;
}

__device__ __forceinline__ float sl1(float d) {
  d = fabsf(d);
  return (d < 1.0f) ? 0.5f * d * d : d - 0.5f;
}

// ---------------- Kernel 2: filter -> sort -> NMS -> per-class argmax -> loss ----------------
__global__ __launch_bounds__(T) void nms_loss_kernel(const float* __restrict__ loc,
                                                     const float* __restrict__ conf,
                                                     const float* __restrict__ tb,
                                                     const int* __restrict__ lbl,
                                                     const float* __restrict__ smax,
                                                     const float* __restrict__ s0,
                                                     float* __restrict__ out) {
  extern __shared__ unsigned char smem[];
  unsigned long long* keys = (unsigned long long*)smem;        // PAD
  int*      fidx    = (int*)(keys + PAD);                      // N_ANCHORS
  float*    sPx     = (float*)(fidx + N_ANCHORS);              // PAD
  float*    sPy     = sPx + PAD;                               // PAD
  int*      kAnchor = (int*)(sPy + PAD);                       // PAD
  unsigned* supp    = (unsigned*)(kAnchor + PAD);              // PAD/32
  int*      scan    = (int*)(supp + PAD / 32);                 // T
  float*    redVals = (float*)(scan + T);                      // 128
  float*    partialArr = redVals + 128;                        // 16
  __shared__ int   sNumPos;
  __shared__ float sTop0;

  const int tid = threadIdx.x;
  const float tb0 = tb[0], tb1 = tb[1], tb2 = tb[2], tb3 = tb[3];

  // ---- Phase A: ordered compaction of anchors with smax > 0.5 ----
  int flags = 0, cnt = 0;
#pragma unroll
  for (int k = 0; k < N_ANCHORS / T; ++k) {
    int a = tid * (N_ANCHORS / T) + k;
    bool ps = smax[a] > 0.5f;
    flags |= ((int)ps) << k;
    cnt += (int)ps;
  }
  int F;
  int base = block_scan(scan, tid, cnt, &F);
#pragma unroll
  for (int k = 0; k < N_ANCHORS / T; ++k) {
    if ((flags >> k) & 1) {
      int a = tid * (N_ANCHORS / T) + k;
      int p = base++;
      fidx[p] = a;
      unsigned sb = __float_as_uint(s0[a]);           // scores in (0,1): bit order == float order
      keys[p] = ((unsigned long long)sb << 32) | (unsigned)(~p); // stable: ties -> smaller p first
    }
  }
  for (int i = tid; i < PAD; i += T)
    if (i >= F) keys[i] = 0ull;                        // pads sort to the end (descending)
  __syncthreads();

  if (F > 0) {
    // ---- Phase B: descending bitonic sort of PAD u64 keys ----
    for (unsigned kk = 2; kk <= PAD; kk <<= 1) {
      for (unsigned j = kk >> 1; j > 0; j >>= 1) {
        for (unsigned i = tid; i < PAD; i += T) {
          unsigned ixj = i ^ j;
          if (ixj > i) {
            unsigned long long x = keys[i], y = keys[ixj];
            bool desc = ((i & kk) == 0);
            if (desc ? (x < y) : (x > y)) { keys[i] = y; keys[ixj] = x; }
          }
        }
        __syncthreads();
      }
    }

    // ---- unpack sorted boxes into LDS ----
    for (int t = tid; t < PAD; t += T) {
      if (t < F) {
        int p = (int)(~(unsigned)(keys[t] & 0xffffffffull)) & (PAD - 1);
        int a = fidx[p];
        sPx[t] = loc[2 * a];
        sPy[t] = loc[2 * a + 1];
      }
    }
    for (int w = tid; w < PAD / 32; w += T) supp[w] = 0u;
    if (tid == 0) sNumPos = 0;
    __syncthreads();

    // ---- Phase C: serial NMS over sorted order (IoU > 0.5 suppresses later boxes) ----
    for (int i = 0; i < F; ++i) {
      unsigned sw = supp[i >> 5];
      if (!((sw >> (i & 31)) & 1u)) {
        float pxi = sPx[i], pyi = sPy[i];
        float ax1 = tb0 * pxi, ay1 = tb1 * pyi, ax2 = tb2 * pxi, ay2 = tb3 * pyi;
        float aarea = (ax2 - ax1) * (ay2 - ay1);
        for (int j = i + 1 + tid; j < F; j += T) {
          float pxj = sPx[j], pyj = sPy[j];
          float bx1 = tb0 * pxj, by1 = tb1 * pyj, bx2 = tb2 * pxj, by2 = tb3 * pyj;
          float barea = (bx2 - bx1) * (by2 - by1);
          float iw = fmaxf(fminf(ax2, bx2) - fmaxf(ax1, bx1), 0.0f);
          float ih = fmaxf(fminf(ay2, by2) - fmaxf(ay1, by1), 0.0f);
          float inter = iw * ih;
          float iou = inter / (aarea + barea - inter);
          if (iou > 0.5f) atomicOr(&supp[j >> 5], 1u << (j & 31));
        }
      }
      __syncthreads();
    }

    // ---- Phase D: compact kept list (sorted order) + num_pos = sum of filtered positions ----
    int kflags = 0, cnt2 = 0;
#pragma unroll
    for (int k = 0; k < PAD / T; ++k) {
      int t = tid * (PAD / T) + k;
      bool kept = (t < F) && !((supp[t >> 5] >> (t & 31)) & 1u);
      kflags |= ((int)kept) << k;
      cnt2 += (int)kept;
    }
    int K;
    int base2 = block_scan(scan, tid, cnt2, &K);
    int localNP = 0;
#pragma unroll
    for (int k = 0; k < PAD / T; ++k) {
      if ((kflags >> k) & 1) {
        int t = tid * (PAD / T) + k;
        int p = (int)(~(unsigned)(keys[t] & 0xffffffffull)) & (PAD - 1);
        kAnchor[base2++] = fidx[p];
        localNP += p;
      }
    }
    atomicAdd(&sNumPos, localNP);
    __syncthreads();

    if (K > 0) {
      // ---- Phase E: per-class max/argmax over kept rows (k=1 top_k) + smooth-L1 per class ----
      if (tid < 128) redVals[tid] = 0.0f;
      if (tid < NCLS) {
        float best = -3.402823466e38f;
        int bi = 0;
        for (int t = 0; t < K; ++t) {
          float v = conf[(size_t)kAnchor[t] * NCLS + tid];  // coalesced across classes
          if (v > best) { best = v; bi = t; }
        }
        int a = kAnchor[bi];
        float bpx = loc[2 * a], bpy = loc[2 * a + 1];
        float l = sl1(tb0 * bpx - tb0) + sl1(tb1 * bpy - tb1) +
                  sl1(tb2 * bpx - tb2) + sl1(tb3 * bpy - tb3);
        redVals[tid] = l;
        if (tid == 0) sTop0 = best;
      }
      __syncthreads();

      // ---- Phase F: WMMA reduction of the 128-slot partial-sum tile (wave 0, EXEC all ones) ----
#if __has_builtin(__builtin_amdgcn_wmma_f32_16x16x4_f32)
      if (tid < 32) {
        if (tid < 16) partialArr[tid] = 0.0f;
        v8f acc = {0.f, 0.f, 0.f, 0.f, 0.f, 0.f, 0.f, 0.f};
#pragma unroll
        for (int tile = 0; tile < 2; ++tile) {
          v2f a; a[0] = redVals[tile * 64 + tid * 2]; a[1] = redVals[tile * 64 + tid * 2 + 1];
          v2f b; b[0] = 1.0f; b[1] = 1.0f;            // B = ones -> D rows hold row-sums of A
          acc = __builtin_amdgcn_wmma_f32_16x16x4_f32(false, a, false, b,
                                                      (short)0, acc, false, false);
        }
        float s = acc[0] + acc[1] + acc[2] + acc[3] + acc[4] + acc[5] + acc[6] + acc[7];
        if (tid == 0)  partialArr[0] = s;   // rows 0..7  (N=0 column, lanes 0..15 half)
        if (tid == 16) partialArr[1] = s;   // rows 8..15 (lanes 16..31 half)
      }
#else
      if (tid < 32) {
        if (tid < 16) partialArr[tid] = 0.0f;
        v16h ones, bv;
#pragma unroll
        for (int h = 0; h < 16; ++h) ones[h] = (_Float16)1.0f;
#pragma unroll
        for (int h = 0; h < 16; ++h) {
          int idx = tid * 16 + h;
          bv[h] = (idx < 128) ? (_Float16)redVals[idx] : (_Float16)0.0f;
        }
        v8f acc = {0.f, 0.f, 0.f, 0.f, 0.f, 0.f, 0.f, 0.f};
        acc = __builtin_amdgcn_wmma_f32_16x16x32_f16(false, ones, false, bv,
                                                     (short)0, acc, false, false);
        if (tid < 16) partialArr[tid] = acc[0];  // D row 0 = per-column sums; sum over N
      }
#endif
      __syncthreads();

      if (tid == 0) {
        float loc_loss = 0.0f;
#pragma unroll
        for (int i = 0; i < 16; ++i) loc_loss += partialArr[i];
        float pred_bin = (sTop0 > 0.5f) ? 1.0f : 0.0f;
        (void)pred_bin;                            // log_softmax of 1-element vector == 0
        float ce = -((float)lbl[0]) * 0.0f;        // == 0
        float t1 = 1.0f - expf(-ce);
        float focal = 0.25f * t1 * t1 * ce;        // == 0, kept for fidelity
        out[0] = (loc_loss + focal) / (float)sNumPos;
      }
    } else {
      if (tid == 0) out[0] = 0.001f;
    }
  } else {
    if (tid == 0) out[0] = 0.001f;
  }
}

// ---------------- Host launcher ----------------
extern "C" void kernel_launch(void* const* d_in, const int* in_sizes, int n_in,
                              void* d_out, int out_size, void* d_ws, size_t ws_size,
                              hipStream_t stream) {
  (void)in_sizes; (void)n_in; (void)out_size; (void)ws_size;
  const float* loc  = (const float*)d_in[0];   // (1, 6144, 2)
  const float* conf = (const float*)d_in[1];   // (1, 6144, 80)
  const float* tbx  = (const float*)d_in[2];   // (1, 1, 4)
  const int*   lbl  = (const int*)d_in[3];     // (1,)
  float* out = (float*)d_out;

  float* smax = (float*)d_ws;
  float* s0   = smax + N_ANCHORS;

  score_kernel<<<dim3(N_ANCHORS / 256), dim3(256), 0, stream>>>(conf, smax, s0);

  // dynamic LDS: keys(64K) + fidx(24K) + sPx/sPy(64K) + kAnchor(32K) + supp(1K)
  //            + scan(4K) + redVals/partial(~0.6K) = 194112 B  (WGP has 320KB)
  const size_t smem_bytes =
      (size_t)PAD * 8 + (size_t)N_ANCHORS * 4 + (size_t)PAD * 4 * 3 +
      (size_t)(PAD / 32) * 4 + (size_t)T * 4 + (size_t)144 * 4;

  nms_loss_kernel<<<dim3(1), dim3(T), smem_bytes, stream>>>(loc, conf, tbx, lbl,
                                                            smax, s0, out);
}
